// AdvancedCBFNet_8452495639041
// MI455X (gfx1250) — matrix operand order repaired and checked by reading, not compiled, not run on hardware
//
#include <hip/hip_runtime.h>
#include <hip/hip_bf16.h>
#include <math.h>

// ---------------------------------------------------------------------------
// AdvancedCBFNet GNN on MI455X (gfx1250).
// Dense layers via v_wmma_f32_16x16x32_f16 with:
//  - A fragments loaded straight from row-major global memory as 2x b128/lane
//  - B (weights) pre-packed into per-lane fragment order (1 contiguous 32B/lane)
//  - each wave: 2 M-tiles x 4 N-tiles -> 8 WMMAs per k-step, B frags reused 2x
//  - OOB rows handled by address clamping only (results never stored), so the
//    WMMA pipe runs with zero per-iteration masking overhead
// ---------------------------------------------------------------------------

typedef _Float16 half_t;
typedef __attribute__((ext_vector_type(16))) _Float16 v16h;
typedef __attribute__((ext_vector_type(8)))  _Float16 v8h;
typedef __attribute__((ext_vector_type(8)))  float    v8f;

#define D_NODE 128
#define D_EDGE 16

__device__ __forceinline__ float swishf(float x) {
  return x / (1.0f + __expf(-x));
}
// order-preserving float <-> uint map for atomicMax-based segment max
__device__ __forceinline__ unsigned fmap(float f) {
  unsigned u = __float_as_uint(f);
  return (u & 0x80000000u) ? ~u : (u | 0x80000000u);
}
__device__ __forceinline__ float funmap(unsigned u) {
  unsigned v = (u & 0x80000000u) ? (u & 0x7FFFFFFFu) : ~u;
  return __uint_as_float(v);
}

// ---------------------------------------------------------------------------
// WMMA GEMM: C[M,N] = act(A[M,K] * B[K,N] + bias) (+ rs * resid)
// A: row-major f16 (lda, 16B-aligned rows). Bp: weights pre-packed in
// B-fragment order (see pack_b_f16). Block: 256 thr = 8 waves; wave w owns
// rows [bm + 32w, +32) as two 16-row M-tiles and all 4 N-tiles of BN=64.
// K mult of 32, N mult of 64.
// ---------------------------------------------------------------------------
#define BM 256
#define BN 64

__global__ __launch_bounds__(256) void gemm_wmma_f16(
    const half_t* __restrict__ A, int lda,
    const half_t* __restrict__ Bp,
    const float*  __restrict__ bias,
    half_t* __restrict__ C, int ldc,
    const half_t* __restrict__ resid, int ldr, float resid_scale,
    int M, int N, int K, int act)
{
  const int tid  = threadIdx.x;
  const int wid  = tid >> 5;
  const int lane = tid & 31;
  const int hi   = lane >> 4;       // lane half: 0 | 1
  const int l16  = lane & 15;

  const int bm = blockIdx.x * BM;
  const int bn = blockIdx.y * BN;

  // this lane's two A rows (clamped; OOB rows compute garbage, never stored)
  const int grow0 = bm + (wid << 5) + l16;
  const int grow1 = grow0 + 16;
  const half_t* Arow0 = A + (size_t)(grow0 < M ? grow0 : (M - 1)) * lda;
  const half_t* Arow1 = A + (size_t)(grow1 < M ? grow1 : (M - 1)) * lda;

  const int ntiles = N >> 4;
  const int nt0    = bn >> 4;

  v8f acc0[4] = {{}, {}, {}, {}};
  v8f acc1[4] = {{}, {}, {}, {}};

  for (int k0 = 0; k0 < K; k0 += 32) {
    // A fragments: two contiguous 8-half (16B) runs of each lane's row.
    // hi=0: K=[k0,k0+8) + [k0+16,k0+24); hi=1: K=[k0+8,+8) + [k0+24,+8)
    v8h a0lo = *(const v8h*)(Arow0 + k0 + (hi << 3));
    v8h a0hi = *(const v8h*)(Arow0 + k0 + 16 + (hi << 3));
    v8h a1lo = *(const v8h*)(Arow1 + k0 + (hi << 3));
    v8h a1hi = *(const v8h*)(Arow1 + k0 + 16 + (hi << 3));
    v16h af0 = __builtin_shufflevector(a0lo, a0hi, 0, 1, 2, 3, 4, 5, 6, 7,
                                       8, 9, 10, 11, 12, 13, 14, 15);
    v16h af1 = __builtin_shufflevector(a1lo, a1hi, 0, 1, 2, 3, 4, 5, 6, 7,
                                       8, 9, 10, 11, 12, 13, 14, 15);

    // B fragments: per-lane contiguous 32B in packed layout
    const half_t* bt = Bp + (((size_t)(k0 >> 5) * ntiles + nt0) << 9) + (lane << 4);
    v16h bf[4];
    #pragma unroll
    for (int nt = 0; nt < 4; ++nt)
      bf[nt] = *(const v16h*)(bt + (nt << 9));

    if (k0 + 32 < K) {
      __builtin_prefetch(Arow0 + k0 + 32 + (hi << 3), 0, 1);
      __builtin_prefetch(Arow1 + k0 + 32 + (hi << 3), 0, 1);
    }

    #pragma unroll
    for (int nt = 0; nt < 4; ++nt) {
      acc0[nt] = __builtin_amdgcn_wmma_f32_16x16x32_f16(
          false, af0, false, bf[nt], (short)0, acc0[nt], false, false);
      acc1[nt] = __builtin_amdgcn_wmma_f32_16x16x32_f16(
          false, af1, false, bf[nt], (short)0, acc1[nt], false, false);
    }
  }

  // C/D layout: lane -> N = lane%16; VGPR r -> M = r (+8 if hi)
  #pragma unroll
  for (int mt = 0; mt < 2; ++mt) {
    #pragma unroll
    for (int nt = 0; nt < 4; ++nt) {
      v8f acc = mt ? acc1[nt] : acc0[nt];
      const int col = bn + (nt << 4) + l16;
      const float bv = bias ? bias[col] : 0.0f;
      #pragma unroll
      for (int r = 0; r < 8; ++r) {
        int row = bm + (wid << 5) + (mt << 4) + r + (hi ? 8 : 0);
        if (row < M) {
          float c = acc[r] + bv;
          if (act == 1) c = swishf(c);
          if (resid) c += resid_scale * (float)resid[(size_t)row * ldr + col];
          C[(size_t)row * ldc + col] = (half_t)c;
        }
      }
    }
  }
}

// ---------------------------------------------------------------------------
// Pack f32 weights (K x N row-major) into f16 WMMA B-fragment order with
// zero-padded K rows. Layout: tile (kt, nt) -> 512 halves; within a tile,
// lane l's 16 halves are contiguous: half h holds
// B[kt*32 + (h&7) + 8*(l>=16) + 16*(h>=8)][nt*16 + l%16].
// ---------------------------------------------------------------------------
__global__ void pack_b_f16(const float* __restrict__ src, half_t* __restrict__ dst,
                           int K, int Kp, int N)
{
  size_t idx = (size_t)blockIdx.x * blockDim.x + threadIdx.x;
  size_t total = (size_t)Kp * N;
  if (idx >= total) return;
  int h    = (int)(idx & 15);
  int lane = (int)((idx >> 4) & 31);
  size_t tile = idx >> 9;
  int nt = (int)(tile % (size_t)(N >> 4));
  int kt = (int)(tile / (size_t)(N >> 4));
  int v = h >> 1, odd = h & 1;
  int k   = kt * 32 + (h & 7) + ((lane >> 4) ? 8 : 0) + ((h >> 3) ? 16 : 0);
  int col = nt * 16 + (lane & 15);
  dst[idx] = (k < K) ? (half_t)src[(size_t)k * N + col] : (half_t)0.0f;
}

// f32 -> f16 plain conversion (row-major)
__global__ void cvt_f16(const float* __restrict__ src, half_t* __restrict__ dst,
                        size_t total)
{
  size_t idx = (size_t)blockIdx.x * blockDim.x + threadIdx.x;
  if (idx >= total) return;
  dst[idx] = (half_t)src[idx];
}

// Build edge rows: [edge(16) | x[sender](dx) | x[receiver](dx) | 0-pad]
__global__ void gather_concat_edges(const float* __restrict__ edges,
                                    const half_t* __restrict__ x, int dx,
                                    const int* __restrict__ snd,
                                    const int* __restrict__ rcv,
                                    half_t* __restrict__ Aout, int lda, int E_)
{
  size_t idx = (size_t)blockIdx.x * blockDim.x + threadIdx.x;
  size_t total = (size_t)E_ * lda;
  if (idx >= total) return;
  int e = (int)(idx / lda);
  int k = (int)(idx % lda);
  float v = 0.0f;
  if (k < D_EDGE)               v = edges[(size_t)e * D_EDGE + k];
  else if (k < D_EDGE + dx)     v = (float)x[(size_t)snd[e] * dx + (k - D_EDGE)];
  else if (k < D_EDGE + 2 * dx) v = (float)x[(size_t)rcv[e] * dx + (k - D_EDGE - dx)];
  Aout[idx] = (half_t)v;
}

// Fuse gate layer 2 (256->128 linear) with attn (128->1 linear):
// wfuse[i] = sum_j Wg1[i,j]*wa[j] ; wfuse[256] = b_g1 . wa + b_attn
__global__ void fuse_attn(const float* __restrict__ g1w, const float* __restrict__ g1b,
                          const float* __restrict__ aw,  const float* __restrict__ ab,
                          float* __restrict__ wfuse)
{
  int i = threadIdx.x;  // 256 threads
  float s = 0.0f;
  for (int j = 0; j < 128; ++j) s += g1w[(size_t)i * 128 + j] * aw[j];
  wfuse[i] = s;
  if (i == 0) {
    float b = ab[0];
    for (int j = 0; j < 128; ++j) b += g1b[j] * aw[j];
    wfuse[256] = b;
  }
}

// Per-edge score + segment max (monotone-uint atomicMax on receivers)
__global__ void edge_scores(const half_t* __restrict__ g1, const float* __restrict__ wfuse,
                            const int* __restrict__ rcv, float* __restrict__ scores,
                            unsigned* __restrict__ maxbuf, int E_)
{
  int e = blockIdx.x * blockDim.x + threadIdx.x;
  if (e >= E_) return;
  const half_t* g = g1 + (size_t)e * 256;
  float s = wfuse[256];
  for (int i = 0; i < 256; ++i) s += (float)g[i] * wfuse[i];
  scores[e] = s;
  atomicMax(&maxbuf[rcv[e]], fmap(s));
}

__global__ void edge_exp(const float* __restrict__ scores, const unsigned* __restrict__ maxbuf,
                         const int* __restrict__ rcv, float* __restrict__ expv,
                         float* __restrict__ denom, int E_)
{
  int e = blockIdx.x * blockDim.x + threadIdx.x;
  if (e >= E_) return;
  int r = rcv[e];
  float ex = __expf(scores[e] - funmap(maxbuf[r]));
  expv[e] = ex;
  atomicAdd(&denom[r], ex);
}

// aggr[recv] += softmax_weight * msgs[e]   (4 channels / thread)
__global__ void edge_aggregate(const half_t* __restrict__ msgs, const float* __restrict__ expv,
                               const float* __restrict__ denom, const int* __restrict__ rcv,
                               float* __restrict__ aggr, int E_)
{
  size_t idx = (size_t)blockIdx.x * blockDim.x + threadIdx.x;
  size_t total = (size_t)E_ * 64;
  if (idx >= total) return;
  int e  = (int)(idx >> 6);
  int c0 = (int)(idx & 63) << 2;
  int r  = rcv[e];
  float w = expv[e] / denom[r];
  #pragma unroll
  for (int j = 0; j < 4; ++j)
    atomicAdd(&aggr[(size_t)r * 256 + c0 + j],
              w * (float)msgs[(size_t)e * 256 + c0 + j]);
}

// Build node rows: [x(dx) | aggr(256)]
__global__ void concat_nodes(const half_t* __restrict__ x, int dx,
                             const float* __restrict__ aggr,
                             half_t* __restrict__ out, int Ku, int Nn)
{
  size_t idx = (size_t)blockIdx.x * blockDim.x + threadIdx.x;
  size_t total = (size_t)Nn * Ku;
  if (idx >= total) return;
  int n = (int)(idx / Ku);
  int k = (int)(idx % Ku);
  float v = (k < dx) ? (float)x[(size_t)n * dx + k]
                     : aggr[(size_t)n * 256 + (k - dx)];
  out[idx] = (half_t)v;
}

// Head MLP on node 0: 256->512->256->128->1, swish hidden, tanh out. M=1:
// plain VALU; WMMA pointless at M=1.
__global__ void head_kernel(const half_t* __restrict__ x,
                            const float* __restrict__ w0, const float* __restrict__ b0,
                            const float* __restrict__ w1, const float* __restrict__ b1,
                            const float* __restrict__ w2, const float* __restrict__ b2,
                            const float* __restrict__ w3, const float* __restrict__ b3,
                            float* __restrict__ out)
{
  __shared__ float h0[256], h1[512], h2[256], h3[128];
  int t = threadIdx.x;  // 256 threads
  h0[t] = (float)x[t];
  __syncthreads();
  for (int o = t; o < 512; o += 256) {
    float s = b0[o];
    for (int i = 0; i < 256; ++i) s += h0[i] * w0[(size_t)i * 512 + o];
    h1[o] = swishf(s);
  }
  __syncthreads();
  {
    float s = b1[t];
    for (int i = 0; i < 512; ++i) s += h1[i] * w1[(size_t)i * 256 + t];
    h2[t] = swishf(s);
  }
  __syncthreads();
  if (t < 128) {
    float s = b2[t];
    for (int i = 0; i < 256; ++i) s += h2[i] * w2[(size_t)i * 128 + t];
    h3[t] = swishf(s);
  }
  __syncthreads();
  if (t == 0) {
    float s = b3[0];
    for (int i = 0; i < 128; ++i) s += h3[i] * w3[i];
    out[0] = tanhf(s);
  }
}

// ---------------------------------------------------------------------------
// Host orchestration.
// Param flatten order (JAX pytree, dict keys sorted):
//  [0]=nodes [1]=edges [2]=senders [3]=receivers
//  head:   4..11  (b,w) x 4
//  layerL: base=12+18*L : attn(b,w) gate[(b,w)x2] msg[(b,w)x3] upd[(b,w)x3]
// ---------------------------------------------------------------------------
extern "C" void kernel_launch(void* const* d_in, const int* in_sizes, int n_in,
                              void* d_out, int out_size, void* d_ws, size_t ws_size,
                              hipStream_t stream) {
  (void)n_in; (void)out_size; (void)ws_size;
  const float* nodes     = (const float*)d_in[0];
  const float* edges     = (const float*)d_in[1];
  const int*   senders   = (const int*)d_in[2];
  const int*   receivers = (const int*)d_in[3];
  const int Nn = in_sizes[0] / D_NODE;
  const int Ee = in_sizes[2];

  const float* hb0 = (const float*)d_in[4];  const float* hw0 = (const float*)d_in[5];
  const float* hb1 = (const float*)d_in[6];  const float* hw1 = (const float*)d_in[7];
  const float* hb2 = (const float*)d_in[8];  const float* hw2 = (const float*)d_in[9];
  const float* hb3 = (const float*)d_in[10]; const float* hw3 = (const float*)d_in[11];

  // workspace carve (~560 MB peak)
  char* p = (char*)d_ws;
  auto carve = [&](size_t bytes) {
    void* r = (void*)p; p += (bytes + 255) & ~(size_t)255; return r;
  };
  half_t* xcur = (half_t*)carve((size_t)Nn * 256 * sizeof(half_t));
  half_t* xnxt = (half_t*)carve((size_t)Nn * 256 * sizeof(half_t));
  half_t* bufA = (half_t*)carve((size_t)Ee * 544 * sizeof(half_t));  // edge concat
  half_t* bufB = (half_t*)carve((size_t)Ee * 512 * sizeof(half_t));  // msg h1
  half_t* bufC = (half_t*)carve((size_t)Ee * 256 * sizeof(half_t));  // msg h2 / gate g1
  half_t* bufD = (half_t*)carve((size_t)Ee * 256 * sizeof(half_t));  // msgs
  half_t* ncat = (half_t*)carve((size_t)Nn * 512 * sizeof(half_t));
  half_t* nh1  = (half_t*)carve((size_t)Nn * 512 * sizeof(half_t));
  half_t* nh2  = (half_t*)carve((size_t)Nn * 256 * sizeof(half_t));
  float*  scores = (float*)carve((size_t)Ee * sizeof(float));
  float*  expv   = (float*)carve((size_t)Ee * sizeof(float));
  unsigned* maxbuf = (unsigned*)carve((size_t)Nn * sizeof(unsigned));
  float*  denom  = (float*)carve((size_t)Nn * sizeof(float));
  float*  aggr   = (float*)carve((size_t)Nn * 256 * sizeof(float));
  float*  wfuse  = (float*)carve(257 * sizeof(float));
  half_t* wm0 = (half_t*)carve((size_t)544 * 512 * sizeof(half_t));
  half_t* wm1 = (half_t*)carve((size_t)512 * 256 * sizeof(half_t));
  half_t* wm2 = (half_t*)carve((size_t)256 * 256 * sizeof(half_t));
  half_t* wg0 = (half_t*)carve((size_t)256 * 256 * sizeof(half_t));
  half_t* wu0 = (half_t*)carve((size_t)512 * 512 * sizeof(half_t));
  half_t* wu1 = (half_t*)carve((size_t)512 * 256 * sizeof(half_t));
  half_t* wu2 = (half_t*)carve((size_t)256 * 256 * sizeof(half_t));

  auto cdiv = [](long long a, long long b) { return (int)((a + b - 1) / b); };
  dim3 blk(256);

  // x0 = f16(nodes)
  cvt_f16<<<cdiv((long long)Nn * D_NODE, 256), blk, 0, stream>>>(
      nodes, xcur, (size_t)Nn * D_NODE);

  int dx = D_NODE;
  for (int l = 0; l < 4; ++l) {
    const int base = 12 + 18 * l;
    const float* attn_b = (const float*)d_in[base + 0];
    const float* attn_w = (const float*)d_in[base + 1];
    const float* g0b = (const float*)d_in[base + 2];
    const float* g0w = (const float*)d_in[base + 3];
    const float* g1b = (const float*)d_in[base + 4];
    const float* g1w = (const float*)d_in[base + 5];
    const float* m0b = (const float*)d_in[base + 6];
    const float* m0w = (const float*)d_in[base + 7];
    const float* m1b = (const float*)d_in[base + 8];
    const float* m1w = (const float*)d_in[base + 9];
    const float* m2b = (const float*)d_in[base + 10];
    const float* m2w = (const float*)d_in[base + 11];
    const float* u0b = (const float*)d_in[base + 12];
    const float* u0w = (const float*)d_in[base + 13];
    const float* u1b = (const float*)d_in[base + 14];
    const float* u1w = (const float*)d_in[base + 15];
    const float* u2b = (const float*)d_in[base + 16];
    const float* u2w = (const float*)d_in[base + 17];

    const int Kc  = D_EDGE + 2 * dx;        // 272 | 528
    const int Kcp = (Kc + 31) & ~31;        // 288 | 544
    const int Ku  = dx + 256;               // 384 | 512 (mult of 32)

    // pack weights to f16 B-fragment order (zero-padded K rows)
    pack_b_f16<<<cdiv((long long)Kcp * 512, 256), blk, 0, stream>>>(m0w, wm0, Kc, Kcp, 512);
    pack_b_f16<<<cdiv(512LL * 256, 256), blk, 0, stream>>>(m1w, wm1, 512, 512, 256);
    pack_b_f16<<<cdiv(256LL * 256, 256), blk, 0, stream>>>(m2w, wm2, 256, 256, 256);
    pack_b_f16<<<cdiv(256LL * 256, 256), blk, 0, stream>>>(g0w, wg0, 256, 256, 256);
    pack_b_f16<<<cdiv((long long)Ku * 512, 256), blk, 0, stream>>>(u0w, wu0, Ku, Ku, 512);
    pack_b_f16<<<cdiv(512LL * 256, 256), blk, 0, stream>>>(u1w, wu1, 512, 512, 256);
    pack_b_f16<<<cdiv(256LL * 256, 256), blk, 0, stream>>>(u2w, wu2, 256, 256, 256);
    fuse_attn<<<1, 256, 0, stream>>>(g1w, g1b, attn_w, attn_b, wfuse);

    hipMemsetAsync(maxbuf, 0, (size_t)Nn * 4, stream);
    hipMemsetAsync(denom, 0, (size_t)Nn * 4, stream);
    hipMemsetAsync(aggr, 0, (size_t)Nn * 256 * 4, stream);

    gather_concat_edges<<<cdiv((long long)Ee * Kcp, 256), blk, 0, stream>>>(
        edges, xcur, dx, senders, receivers, bufA, Kcp, Ee);

    // msg MLP: Kcp->512 (swish) ->256 (swish) ->256 (linear)
    gemm_wmma_f16<<<dim3(cdiv(Ee, BM), 512 / BN), blk, 0, stream>>>(
        bufA, Kcp, wm0, m0b, bufB, 512, nullptr, 0, 0.f, Ee, 512, Kcp, 1);
    gemm_wmma_f16<<<dim3(cdiv(Ee, BM), 256 / BN), blk, 0, stream>>>(
        bufB, 512, wm1, m1b, bufC, 256, nullptr, 0, 0.f, Ee, 256, 512, 1);
    gemm_wmma_f16<<<dim3(cdiv(Ee, BM), 256 / BN), blk, 0, stream>>>(
        bufC, 256, wm2, m2b, bufD, 256, nullptr, 0, 0.f, Ee, 256, 256, 0);

    // gate layer 1 (swish) then fused (gate2 ∘ attn) scores + segment softmax
    gemm_wmma_f16<<<dim3(cdiv(Ee, BM), 256 / BN), blk, 0, stream>>>(
        bufD, 256, wg0, g0b, bufC, 256, nullptr, 0, 0.f, Ee, 256, 256, 1);
    edge_scores<<<cdiv(Ee, 256), blk, 0, stream>>>(bufC, wfuse, receivers, scores, maxbuf, Ee);
    edge_exp<<<cdiv(Ee, 256), blk, 0, stream>>>(scores, maxbuf, receivers, expv, denom, Ee);
    edge_aggregate<<<cdiv((long long)Ee * 64, 256), blk, 0, stream>>>(
        bufD, expv, denom, receivers, aggr, Ee);

    // node update MLP: [x|aggr](Ku)->512 (swish) ->256 (swish) ->256 (+0.5x)
    concat_nodes<<<cdiv((long long)Nn * Ku, 256), blk, 0, stream>>>(
        xcur, dx, aggr, ncat, Ku, Nn);
    gemm_wmma_f16<<<dim3(cdiv(Nn, BM), 512 / BN), blk, 0, stream>>>(
        ncat, Ku, wu0, u0b, nh1, 512, nullptr, 0, 0.f, Nn, 512, Ku, 1);
    gemm_wmma_f16<<<dim3(cdiv(Nn, BM), 256 / BN), blk, 0, stream>>>(
        nh1, 512, wu1, u1b, nh2, 256, nullptr, 0, 0.f, Nn, 256, 512, 1);
    const half_t* resid = (l > 0) ? xcur : nullptr;  // dims match only for l>=1
    gemm_wmma_f16<<<dim3(cdiv(Nn, BM), 256 / BN), blk, 0, stream>>>(
        nh2, 256, wu2, u2b, xnxt, 256, resid, dx, 0.5f, Nn, 256, 256, 0);

    half_t* t = xcur; xcur = xnxt; xnxt = t;
    dx = 256;
  }

  head_kernel<<<1, 256, 0, stream>>>(xcur, hw0, hb0, hw1, hb1, hw2, hb2, hw3, hb3,
                                     (float*)d_out);
}
